// GCEdecoder_5016521801841
// MI455X (gfx1250) — matrix operand order repaired and checked by reading, not compiled, not run on hardware
//
#include <hip/hip_runtime.h>
#include <hip/hip_bf16.h>

// Problem constants (match reference)
#define BB 128
#define TT 512
#define DD 400
#define NV 1024
#define NA 128
#define DPAD 416          // 13 * 32, K padded for bf16 WMMA (K=32 per op)
#define NK 13             // K-chunks of 32
#define TTILE 64          // t-columns staged per iteration (4 WMMA N-tiles)

typedef __attribute__((ext_vector_type(16))) __bf16 v16bf;
typedef __attribute__((ext_vector_type(4)))  __bf16 v4bf;
typedef __attribute__((ext_vector_type(8)))  float  v8f;

// ---------------------------------------------------------------------------
// Kernel 1: h_w[b,t] = dot(H[b,t,:], W_score)   (one wave32 per row)
// ---------------------------------------------------------------------------
__global__ void hw_kernel(const float* __restrict__ H,
                          const float* __restrict__ W,
                          float* __restrict__ hw) {
    int wave = (blockIdx.x << 3) + (threadIdx.x >> 5);   // 8 waves / block
    int lane = threadIdx.x & 31;
    if (wave >= BB * TT) return;
    const float* row = H + (size_t)wave * DD;
    float acc = 0.f;
    for (int d = lane; d < DD; d += 32) acc += row[d] * W[d];
#pragma unroll
    for (int off = 16; off > 0; off >>= 1) acc += __shfl_xor(acc, off, 32);
    if (lane == 0) hw[wave] = acc;
}

// ---------------------------------------------------------------------------
// Kernel 2: fused  scores = H[b] x vals^T  +  masked online softmax over t
//           + weighted reduce with h_w  ->  y_utts[b,v]
// grid = B * (NV/128) workgroups, 256 threads (8 waves, 16 v-rows per wave).
// Each iteration stages 64 t-rows of H[b] (bf16, K-padded) and runs FOUR
// independent 13-WMMA chains per wave sharing the register-resident A
// fragments: one dscnt wait feeds 4 back-to-back WMMAs. Softmax state is
// PER-LANE (no cross-lane ops in the hot loop); lane states merge at the end.
// ---------------------------------------------------------------------------
__global__ void __launch_bounds__(256)
attn_kernel(const float* __restrict__ H,       // (B,T,D)
            const float* __restrict__ vals,    // (NV,D)  (= C_vals[:,0,:])
            const float* __restrict__ hw,      // (B,T)
            const int*   __restrict__ lens,    // (B,)
            const float* __restrict__ bsc,     // (1,)
            float* __restrict__ y_utts) {      // (B,NV)
    __shared__ __bf16 Hs[TTILE * DPAD];        // 64 t-rows of H[b], bf16 (53KB)

    const int b     = blockIdx.x >> 3;         // 8 v-tiles per b
    const int vtile = blockIdx.x & 7;
    const int wave  = threadIdx.x >> 5;
    const int lane  = threadIdx.x & 31;
    const int vb    = vtile * 128 + wave * 16; // first v-row of this wave
    const int len   = lens[b];
    const float b0  = bsc[0];

    // ---- Load A (vals tile) fragments into registers, f32 -> bf16, K-padded.
    // A 16x32 in this code's contiguous k order (B matches): lanes 0-15 hold M=lane, K = 0..15;
    // lanes 16-31 hold M=lane-16, K = 16..31.  D=400 is 8-aligned,
    // so each 8-group is all-valid or all-pad.
    const float* vrow = vals + (size_t)(vb + (lane & 15)) * DD;
    v16bf afrag[NK];
#pragma unroll
    for (int k = 0; k < NK; ++k) {
        const int g0 = k * 32 + ((lane & 16) ? 16 : 0);
        const int g1 = g0 + 8;
        v16bf a;
#pragma unroll
        for (int j = 0; j < 8; ++j) {
            float f0 = (g0 < DD) ? vrow[g0 + j] : 0.f;
            float f1 = (g1 < DD) ? vrow[g1 + j] : 0.f;
            a[j]     = (__bf16)f0;
            a[j + 8] = (__bf16)f1;
        }
        afrag[k] = a;
    }

    // Per-lane online-softmax state (VGPR r -> v-row r / r+8 of the D tile)
    float m[8], s[8], sw[8];
#pragma unroll
    for (int r = 0; r < 8; ++r) { m[r] = -1e30f; s[r] = 0.f; sw[r] = 0.f; }

    const float* Hb = H + (size_t)b * TT * DD;

    for (int tile = 0; tile < TT / TTILE; ++tile) {
        const int tb = tile * TTILE;

        // ---- Stage 64 x 400 f32 -> 64 x 416 bf16 into LDS (float4 wide)
        for (int g = threadIdx.x; g < TTILE * (DPAD / 4); g += 256) {
            const int row = g / (DPAD / 4);
            const int d   = (g - row * (DPAD / 4)) * 4;
            float4 f = make_float4(0.f, 0.f, 0.f, 0.f);
            if (d < DD)
                f = *(const float4*)&Hb[(size_t)(tb + row) * DD + d];
            v4bf o;
            o[0] = (__bf16)f.x; o[1] = (__bf16)f.y;
            o[2] = (__bf16)f.z; o[3] = (__bf16)f.w;
            *(v4bf*)&Hs[row * DPAD + d] = o;
        }
        // Prefetch next t-tile while we compute (global_prefetch_b8)
        if (tile + 1 < TT / TTILE) {
            const char* nxt = (const char*)(Hb + (size_t)(tb + TTILE) * DD);
            __builtin_prefetch(nxt + threadIdx.x * 400, 0, 1);
        }

        // Issue hw/mask loads early so they overlap the WMMA chains.
        const int  t0 = tb + (lane & 15);
        const int  t1 = t0 + 16, t2 = t0 + 32, t3 = t0 + 48;
        const bool va0 = (t0 < len), va1 = (t1 < len);
        const bool va2 = (t2 < len), va3 = (t3 < len);
        const float hw0 = hw[b * TT + t0];
        const float hw1 = hw[b * TT + t1];
        const float hw2 = hw[b * TT + t2];
        const float hw3 = hw[b * TT + t3];

        __syncthreads();

        // ---- Four independent 16(v) x 16(t) WMMA chains over 13 K-chunks.
        // B (32x16) layout: lane%16 = t-column; lanes 0-15: K 0..15,
        // lanes 16-31: K 16..31 (contiguous 16 bf16 = one 32B aligned read).
        v8f c0 = {}, c1 = {}, c2 = {}, c3 = {};
        const int tr0 = (lane & 15) * DPAD + ((lane & 16) ? 16 : 0);
        const int tr1 = tr0 + 16 * DPAD;
        const int tr2 = tr0 + 32 * DPAD;
        const int tr3 = tr0 + 48 * DPAD;
#pragma unroll
        for (int k = 0; k < NK; ++k) {
            const int ko = k * 32;
            const v16bf bf0 = *(const v16bf*)&Hs[tr0 + ko];
            const v16bf bf1 = *(const v16bf*)&Hs[tr1 + ko];
            const v16bf bf2 = *(const v16bf*)&Hs[tr2 + ko];
            const v16bf bf3 = *(const v16bf*)&Hs[tr3 + ko];
            c0 = __builtin_amdgcn_wmma_f32_16x16x32_bf16(
                     false, afrag[k], false, bf0, (short)0, c0, false, false);
            c1 = __builtin_amdgcn_wmma_f32_16x16x32_bf16(
                     false, afrag[k], false, bf1, (short)0, c1, false, false);
            c2 = __builtin_amdgcn_wmma_f32_16x16x32_bf16(
                     false, afrag[k], false, bf2, (short)0, c2, false, false);
            c3 = __builtin_amdgcn_wmma_f32_16x16x32_bf16(
                     false, afrag[k], false, bf3, (short)0, c3, false, false);
        }

        // ---- Per-lane online softmax update (8 rows x 4 t-columns)
#pragma unroll
        for (int r = 0; r < 8; ++r) {
            const float s0 = va0 ? c0[r] : -1e30f;
            const float s1 = va1 ? c1[r] : -1e30f;
            const float s2 = va2 ? c2[r] : -1e30f;
            const float s3 = va3 ? c3[r] : -1e30f;
            const float mn = fmaxf(fmaxf(m[r], fmaxf(s0, s1)), fmaxf(s2, s3));
            const float corr = __expf(m[r] - mn);
            const float p0 = va0 ? __expf(s0 - mn) : 0.f;
            const float p1 = va1 ? __expf(s1 - mn) : 0.f;
            const float p2 = va2 ? __expf(s2 - mn) : 0.f;
            const float p3 = va3 ? __expf(s3 - mn) : 0.f;
            s[r]  = s[r]  * corr + (p0 + p1) + (p2 + p3);
            sw[r] = sw[r] * corr + (p0 * hw0 + p1 * hw1)
                                 + (p2 * hw2 + p3 * hw3);
            m[r]  = mn;
        }
        __syncthreads();
    }

    // ---- Merge the 16 per-lane softmax states of each row (once), emit y.
#pragma unroll
    for (int r = 0; r < 8; ++r) {
        float mr = m[r], sr = s[r], wr = sw[r];
#pragma unroll
        for (int off = 1; off <= 8; off <<= 1) {
            const float m2 = __shfl_xor(mr, off, 32);
            const float s2 = __shfl_xor(sr, off, 32);
            const float w2 = __shfl_xor(wr, off, 32);
            const float mn = fmaxf(mr, m2);
            const float ca = __expf(mr - mn);
            const float cb = __expf(m2 - mn);
            sr = sr * ca + s2 * cb;
            wr = wr * ca + w2 * cb;
            mr = mn;
        }
        const float y = wr / sr + b0;
        if (lane == 0)  y_utts[b * NV + vb + r]     = y;
        if (lane == 16) y_utts[b * NV + vb + 8 + r] = y;
    }
}

// ---------------------------------------------------------------------------
// Kernel 3: acts path part 1 — s2 = C_acts·c_utt, softmax, q_acts (per b)
// ---------------------------------------------------------------------------
__global__ void acts1_kernel(const float* __restrict__ C_acts, // (B,NA,D)
                             const float* __restrict__ c_utt,  // (B,D)
                             float* __restrict__ qa) {          // (B,D)
    __shared__ float s2[NA];
    __shared__ float p2[NA];
    const int b = blockIdx.x, wave = threadIdx.x >> 5, lane = threadIdx.x & 31;
    const float* cu = c_utt + (size_t)b * DD;

    for (int i = 0; i < NA / 8; ++i) {
        const int a = wave + 8 * i;
        const float* crow = C_acts + ((size_t)b * NA + a) * DD;
        float acc = 0.f;
        for (int d = lane; d < DD; d += 32) acc += crow[d] * cu[d];
#pragma unroll
        for (int off = 16; off > 0; off >>= 1) acc += __shfl_xor(acc, off, 32);
        if (lane == 0) s2[a] = acc;
    }
    __syncthreads();
    if (threadIdx.x < 32) {
        const int l = threadIdx.x;
        float v0 = s2[l], v1 = s2[l + 32], v2 = s2[l + 64], v3 = s2[l + 96];
        float mx = fmaxf(fmaxf(v0, v1), fmaxf(v2, v3));
#pragma unroll
        for (int off = 16; off > 0; off >>= 1) mx = fmaxf(mx, __shfl_xor(mx, off, 32));
        float e0 = __expf(v0 - mx), e1 = __expf(v1 - mx);
        float e2 = __expf(v2 - mx), e3 = __expf(v3 - mx);
        float sum = e0 + e1 + e2 + e3;
#pragma unroll
        for (int off = 16; off > 0; off >>= 1) sum += __shfl_xor(sum, off, 32);
        const float inv = 1.f / sum;
        p2[l] = e0 * inv; p2[l + 32] = e1 * inv;
        p2[l + 64] = e2 * inv; p2[l + 96] = e3 * inv;
    }
    __syncthreads();
    for (int d = threadIdx.x; d < DD; d += 256) {
        float acc = 0.f;
        for (int a = 0; a < NA; ++a)
            acc += p2[a] * C_acts[((size_t)b * NA + a) * DD + d];
        qa[(size_t)b * DD + d] = acc;
    }
}

// ---------------------------------------------------------------------------
// Kernel 4: y_acts[b,v] = dot(q_acts[b,:], vals[v,:])   (one wave per (b,v))
// ---------------------------------------------------------------------------
__global__ void acts2_kernel(const float* __restrict__ qa,
                             const float* __restrict__ vals,
                             float* __restrict__ y_acts) {
    const int gw = (blockIdx.x << 3) + (threadIdx.x >> 5);
    if (gw >= BB * NV) return;
    const int b = gw >> 10, v = gw & (NV - 1), lane = threadIdx.x & 31;
    const float* q  = qa   + (size_t)b * DD;
    const float* vr = vals + (size_t)v * DD;
    float acc = 0.f;
    for (int d = lane; d < DD; d += 32) acc += q[d] * vr[d];
#pragma unroll
    for (int off = 16; off > 0; off >>= 1) acc += __shfl_xor(acc, off, 32);
    if (lane == 0) y_acts[b * NV + v] = acc;
}

// ---------------------------------------------------------------------------
extern "C" void kernel_launch(void* const* d_in, const int* in_sizes, int n_in,
                              void* d_out, int out_size, void* d_ws, size_t ws_size,
                              hipStream_t stream) {
    const float* H      = (const float*)d_in[0];  // (B,T,D)
    const float* c_utt  = (const float*)d_in[1];  // (B,D)
    const float* C_acts = (const float*)d_in[2];  // (B,NA,D)
    const float* C_vals = (const float*)d_in[3];  // (NV,1,D) -> (NV,D)
    const float* W      = (const float*)d_in[4];  // (1,D)
    const float* bsc    = (const float*)d_in[5];  // (1,)
    const int*   lens   = (const int*)d_in[6];    // (B,)

    float* y_utts = (float*)d_out;                // B*NV
    float* y_acts = y_utts + BB * NV;             // B*NV

    float* hw = (float*)d_ws;                     // B*T floats
    float* qa = hw + BB * TT;                     // B*D floats

    // h_w[b,t] = H[b,t,:] . W   (one wave per row; 65536 rows, 8 waves/block)
    hw_kernel<<<(BB * TT) / 8, 256, 0, stream>>>(H, W, hw);

    // fused scores-GEMM + online softmax + h_w reduce -> y_utts
    attn_kernel<<<BB * (NV / 128), 256, 0, stream>>>(H, C_vals, hw, lens, bsc,
                                                     y_utts);

    // acts path
    acts1_kernel<<<BB, 256, 0, stream>>>(C_acts, c_utt, qa);
    acts2_kernel<<<(BB * NV) / 8, 256, 0, stream>>>(qa, C_vals, y_acts);
}
